// AttnWithBias_77412490543864
// MI455X (gfx1250) — compile-verified, matrix operand order checked by reading, not run on hardware
//
#include <hip/hip_runtime.h>

// ---------------------------------------------------------------------------
// AttnWithBias on MI455X (gfx1250, wave32, WMMA + Tensor Data Mover).
//
// Algebraic optimization: einsum 'bqnk,bvnd->bqnd' factorizes into
// (rowsum of softmax) x (sum_v V); the softmax rowsum is identically 1, so
// q/k/bias-network/softmax are dead.  Live work:
//   vsum[b] = (sum_seq pieces[b]) @ Wv + 64*bv
//   s       = LN(state + vsum, norm1)
//   H       = relu(s @ W1 + b1)
//   out     = LN(s + H @ W2 + b2, norm2)
// FFN GEMMs (2 x 412 GFLOP) dominate -> f16 hi/lo-split WMMA GEMM (fp32-level
// accuracy, 3x v_wmma_f32_16x16x32_f16 per K-step).  All GEMM operands are
// pre-split to f16 hi/lo in global memory; tiles are staged purely by TDM
// tensor_load_to_lds (hardware LDS padding) into a double-buffered LDS
// ping-pong so DMA overlaps the WMMA pipe; waves only ds_load + wmma.
// ---------------------------------------------------------------------------

typedef __attribute__((ext_vector_type(16))) _Float16 v16h;
typedef __attribute__((ext_vector_type(8)))  float    v8f;
typedef unsigned __attribute__((ext_vector_type(4)))  v4u;
typedef int      __attribute__((ext_vector_type(4)))  v4i;
typedef int      __attribute__((ext_vector_type(8)))  v8i;

#define B_    256
#define LEN1_ 192
#define LEN2_ 64
#define D_    1024
#define FF_   4096
#define EPS_  1e-5f

#define BM 128
#define BN 128
#define BK 32
#define LDSA (BK + 8)       // halves: 64B row + 16B pad (matches TDM pad fields)
#define TILE_H (BM * LDSA)  // halves per LDS tile

union FragAB { v16h v; _Float16 h[16]; unsigned u[8]; };
union FragC  { v8f  v; float   f[8]; };

// ---------------------------------------------------------------------------
// TDM 2D tile load: 128 rows x 32 halves, row stride K halves, into LDS with
// hardware padding of 4 DWORDs after every 16 DWORDs (-> LDSA=40 layout).
// D# per CDNA5 ISA ch.8; 6-arg therock builtin (g0,g1,g2,g3,g4,cpol).
// ---------------------------------------------------------------------------
__device__ __forceinline__ void tdm_load_tile(unsigned lds_byte_addr,
                                              const void* gptr,
                                              unsigned K_elems,
                                              unsigned rows_total) {
    union { unsigned long long u64; unsigned u32[2]; } ga;
    ga.u64 = (unsigned long long)(__SIZE_TYPE__)gptr;

    v4u g0;
    g0.x = 1u;                                        // count=1 (valid), user mode
    g0.y = lds_byte_addr;                             // lds_addr
    g0.z = ga.u32[0];                                 // global_addr[31:0]
    g0.w = (ga.u32[1] & 0x01FFFFFFu) | 0x80000000u;   // global_addr[56:32] | type=2

    const unsigned w0 = (1u << 16)    // data_size = 1 -> 2 bytes
                      | (1u << 20)    // pad_enable
                      | (3u << 22)    // pad_interval = 3 -> every 16 DWORDs (64B)
                      | (3u << 25);   // pad_amount   = 3 -> 4 DWORDs (16B)
    const unsigned w1 = (K_elems & 0xFFFFu) << 16;                 // tensor_dim0 lo
    const unsigned w2 = ((K_elems >> 16) & 0xFFFFu)
                      | ((rows_total & 0xFFFFu) << 16);            // tensor_dim1 lo
    const unsigned w3 = ((rows_total >> 16) & 0xFFFFu)
                      | (32u << 16);                               // tile_dim0 = 32
    const unsigned w4 = 128u;                                      // tile_dim1 = 128
    const unsigned w5 = K_elems;                                   // dim0 stride
    v8i g1;
    g1[0] = (int)w0; g1[1] = (int)w1; g1[2] = (int)w2; g1[3] = (int)w3;
    g1[4] = (int)w4; g1[5] = (int)w5; g1[6] = 0; g1[7] = 0;

    v4i z4 = {0, 0, 0, 0};                            // groups 2,3 unused (2D)
    v8i z8 = {0, 0, 0, 0, 0, 0, 0, 0};
    __builtin_amdgcn_tensor_load_to_lds(g0, g1, z4, z4, z8, 0);
}

// ---------------------------------------------------------------------------
// WMMA fragment gather from LDS tile [rows][LDSA] halves, K-contiguous.
// 16-bit A-matrix 16x32 layout (ISA 7.12.2): lanes 0-15: K {0..7,16..23};
// lanes 16-31: K {8..15,24..31}; contiguous pairs -> ds_load_b128 merges.
// ---------------------------------------------------------------------------
__device__ __forceinline__ FragAB load_frag(const _Float16* tile, int row0, int lane) {
    FragAB f;
    const int r = row0 + (lane & 15);
    const int g = (lane >> 4) ? 8 : 0;
    const _Float16* p = tile + r * LDSA;
#pragma unroll
    for (int i = 0; i < 8; ++i) {
        const int kb = ((i < 4) ? (i * 2) : (16 + (i - 4) * 2)) + g;
        unsigned w;
        __builtin_memcpy(&w, p + kb, 4);
        f.u[i] = w;
    }
    return f;
}

__device__ __forceinline__ void split_f32(float x, _Float16& hi, _Float16& lo) {
    hi = (_Float16)x;
    lo = (_Float16)(x - (float)hi);
}

// ---------------------------------------------------------------------------
// GEMM: C[M,N] = epi( A[M,K] x B[N,K]^T + bias_scale*bias[n] )
// A: row-major [M][K] f16 hi/lo;  B: K-transposed [N][K] f16 hi/lo.
// Double-buffered TDM ping-pong: waves 0-3 each own one tile DMA.
// ---------------------------------------------------------------------------
template <bool RELU, bool W32, bool W16>
__global__ __launch_bounds__(256)
void tdm_gemm_kernel(const _Float16* __restrict__ Ahi, const _Float16* __restrict__ Alo,
                     const _Float16* __restrict__ Bhi, const _Float16* __restrict__ Blo,
                     const float* __restrict__ bias, float bias_scale,
                     float* __restrict__ C32,
                     _Float16* __restrict__ Chi, _Float16* __restrict__ Clo,
                     int M, int N, int K) {
    __shared__ _Float16 smem[8 * TILE_H];   // [buf 0/1][Ah,Al,Bh,Bl][TILE_H] = 80KB

    const int tid  = threadIdx.x;
    const int lane = tid & 31;
    const int wid  = tid >> 5;           // 0..7
    const int wm   = (wid & 1) * 64;
    const int wn   = (wid >> 1) * 32;

    const int bn0 = blockIdx.x * BN;
    const int bm0 = blockIdx.y * BM;

    FragC acc[4][2];
#pragma unroll
    for (int mi = 0; mi < 4; ++mi)
#pragma unroll
        for (int ni = 0; ni < 2; ++ni)
#pragma unroll
            for (int j = 0; j < 8; ++j) acc[mi][ni].f[j] = 0.0f;

    // waves 0-3 each issue one tile DMA per K-step
    auto issue = [&](int buf, int kt) {
        if (wid < 4) {
            const __SIZE_TYPE__ aoff = (__SIZE_TYPE__)bm0 * K + kt;
            const __SIZE_TYPE__ boff = (__SIZE_TYPE__)bn0 * K + kt;
            const _Float16* src = (wid == 0) ? (Ahi + aoff)
                                : (wid == 1) ? (Alo + aoff)
                                : (wid == 2) ? (Bhi + boff)
                                             : (Blo + boff);
            _Float16* dst = smem + (buf * 4 + wid) * TILE_H;
            const unsigned rows = (wid < 2) ? (unsigned)M : (unsigned)N;
            tdm_load_tile((unsigned)(__SIZE_TYPE__)dst, src, (unsigned)K, rows);
        }
    };
    auto drain = [&]() {
        if (wid < 4) __builtin_amdgcn_s_wait_tensorcnt(0);
    };
    auto compute = [&](int buf) {
        const _Float16* cAh = smem + (buf * 4 + 0) * TILE_H;
        const _Float16* cAl = smem + (buf * 4 + 1) * TILE_H;
        const _Float16* cBh = smem + (buf * 4 + 2) * TILE_H;
        const _Float16* cBl = smem + (buf * 4 + 3) * TILE_H;
        FragAB ah[4], al[4], bh[2], bl[2];
#pragma unroll
        for (int mi = 0; mi < 4; ++mi) {
            ah[mi] = load_frag(cAh, wm + mi * 16, lane);
            al[mi] = load_frag(cAl, wm + mi * 16, lane);
        }
#pragma unroll
        for (int ni = 0; ni < 2; ++ni) {
            bh[ni] = load_frag(cBh, wn + ni * 16, lane);
            bl[ni] = load_frag(cBl, wn + ni * 16, lane);
        }
#pragma unroll
        for (int mi = 0; mi < 4; ++mi)
#pragma unroll
            for (int ni = 0; ni < 2; ++ni) {
                acc[mi][ni].v = __builtin_amdgcn_wmma_f32_16x16x32_f16(
                    false, ah[mi].v, false, bh[ni].v, (short)0, acc[mi][ni].v, false, false);
                acc[mi][ni].v = __builtin_amdgcn_wmma_f32_16x16x32_f16(
                    false, ah[mi].v, false, bl[ni].v, (short)0, acc[mi][ni].v, false, false);
                acc[mi][ni].v = __builtin_amdgcn_wmma_f32_16x16x32_f16(
                    false, al[mi].v, false, bh[ni].v, (short)0, acc[mi][ni].v, false, false);
            }
    };

    // prologue: fill buf0 with tile 0
    issue(0, 0);
    drain();
    __syncthreads();

    // K is a multiple of 2*BK; ping-pong buf0/buf1, DMA overlapped with WMMA
    for (int kt = 0; kt < K; kt += 2 * BK) {
        if (kt + BK < K) issue(1, kt + BK);      // prefetch next into buf1
        compute(0);                              // consume buf0
        if (kt + BK < K) drain();
        __syncthreads();

        if (kt + 2 * BK < K) issue(0, kt + 2 * BK);  // prefetch into buf0
        compute(1);                                  // consume buf1
        if (kt + 2 * BK < K) drain();
        __syncthreads();
    }

    // epilogue: C/D layout VGPR j -> (M = j + 8*(lane>=16), N = lane&15)
    const int g8 = (lane >> 4) * 8;
#pragma unroll
    for (int mi = 0; mi < 4; ++mi)
#pragma unroll
        for (int ni = 0; ni < 2; ++ni) {
            const int col = bn0 + wn + ni * 16 + (lane & 15);
            const int row = bm0 + wm + mi * 16 + g8;
            const float bv = bias_scale * bias[col];
#pragma unroll
            for (int j = 0; j < 8; ++j) {
                float o = acc[mi][ni].f[j] + bv;
                if (RELU) o = fmaxf(o, 0.0f);
                const __SIZE_TYPE__ idx = (__SIZE_TYPE__)(row + j) * N + col;
                if (W32) C32[idx] = o;
                if (W16) { _Float16 h, l; split_f32(o, h, l); Chi[idx] = h; Clo[idx] = l; }
            }
        }
}

// ---------------------------------------------------------------------------
// Weight prep: W[K][N] fp32 -> WT_hi/lo[N][K] f16 (transpose via LDS tile)
// ---------------------------------------------------------------------------
__global__ __launch_bounds__(256)
void split_transpose_kernel(const float* __restrict__ W,
                            _Float16* __restrict__ Thi, _Float16* __restrict__ Tlo,
                            int K, int N) {
    __shared__ float tile[32][33];
    const int n0 = blockIdx.x * 32;
    const int k0 = blockIdx.y * 32;
    const int tx = threadIdx.x & 31;
    const int ty = threadIdx.x >> 5;    // 0..7
#pragma unroll
    for (int i = ty; i < 32; i += 8)
        tile[i][tx] = W[(__SIZE_TYPE__)(k0 + i) * N + n0 + tx];
    __syncthreads();
#pragma unroll
    for (int i = ty; i < 32; i += 8) {
        const float x = tile[tx][i];    // = W[k0+tx][n0+i]
        _Float16 h, l; split_f32(x, h, l);
        const __SIZE_TYPE__ idx = (__SIZE_TYPE__)(n0 + i) * K + k0 + tx;
        Thi[idx] = h; Tlo[idx] = l;
    }
}

// ---------------------------------------------------------------------------
// piecesum[b,d] = sum_l pieces[b,l,d]  (written directly as f16 hi/lo)
// ---------------------------------------------------------------------------
__global__ void colsum_kernel(const float* __restrict__ pieces,
                              _Float16* __restrict__ phi, _Float16* __restrict__ plo) {
    const int idx = blockIdx.x * blockDim.x + threadIdx.x;   // b*1024 + d
    const int b = idx >> 10;
    const int d = idx & 1023;
    const float* p = pieces + (__SIZE_TYPE__)b * LEN2_ * D_ + d;
    float s = 0.0f;
#pragma unroll 8
    for (int l = 0; l < LEN2_; ++l) s += p[(__SIZE_TYPE__)l * D_];
    _Float16 h, l2; split_f32(s, h, l2);
    phi[idx] = h; plo[idx] = l2;
}

// ---------------------------------------------------------------------------
// out[row] = LN(x[row] + r[row/rep]) * g + b ; optionally emit f16 hi/lo copy
// ---------------------------------------------------------------------------
__global__ __launch_bounds__(256)
void add_ln_kernel(const float* __restrict__ x, const float* __restrict__ r,
                   const float* __restrict__ gamma, const float* __restrict__ beta,
                   float* __restrict__ out,
                   _Float16* __restrict__ ohi, _Float16* __restrict__ olo,
                   int rep, int do_split) {
    __shared__ float red[256];
    const int row = blockIdx.x;
    const int t = threadIdx.x;
    const float* xr = x + (__SIZE_TYPE__)row * D_;
    const float* rr = r + (__SIZE_TYPE__)(row / rep) * D_;

    float v[4];
    float s = 0.0f;
#pragma unroll
    for (int j = 0; j < 4; ++j) {
        const int i = t + j * 256;
        v[j] = xr[i] + rr[i];
        s += v[j];
    }
    red[t] = s;
    __syncthreads();
    for (int off = 128; off > 0; off >>= 1) {
        if (t < off) red[t] += red[t + off];
        __syncthreads();
    }
    const float mu = red[0] * (1.0f / D_);
    __syncthreads();

    float sv = 0.0f;
#pragma unroll
    for (int j = 0; j < 4; ++j) { const float d = v[j] - mu; sv += d * d; }
    red[t] = sv;
    __syncthreads();
    for (int off = 128; off > 0; off >>= 1) {
        if (t < off) red[t] += red[t + off];
        __syncthreads();
    }
    const float rs = rsqrtf(red[0] * (1.0f / D_) + EPS_);
#pragma unroll
    for (int j = 0; j < 4; ++j) {
        const int i = t + j * 256;
        const float y = (v[j] - mu) * rs * gamma[i] + beta[i];
        const __SIZE_TYPE__ idx = (__SIZE_TYPE__)row * D_ + i;
        out[idx] = y;
        if (do_split) { _Float16 h, l; split_f32(y, h, l); ohi[idx] = h; olo[idx] = l; }
    }
}

// ---------------------------------------------------------------------------
extern "C" void kernel_launch(void* const* d_in, const int* in_sizes, int n_in,
                              void* d_out, int out_size, void* d_ws, size_t ws_size,
                              hipStream_t stream) {
    (void)in_sizes; (void)n_in; (void)out_size; (void)ws_size;

    const float* state = (const float*)d_in[0];
    const float* pieces= (const float*)d_in[1];
    const float* vw    = (const float*)d_in[6];
    const float* vb    = (const float*)d_in[7];
    const float* w1    = (const float*)d_in[8];
    const float* b1    = (const float*)d_in[9];
    const float* w2    = (const float*)d_in[10];
    const float* b2    = (const float*)d_in[11];
    const float* n1g   = (const float*)d_in[12];
    const float* n1b   = (const float*)d_in[13];
    const float* n2g   = (const float*)d_in[14];
    const float* n2b   = (const float*)d_in[15];
    float* out = (float*)d_out;

    const size_t Mr = (size_t)B_ * LEN1_;             // 49152
    // fp32 segments
    float* s   = (float*)d_ws;                        // Mr*1024
    float* Y   = s + Mr * D_;                         // Mr*1024
    float* vs  = Y + Mr * D_;                         // 256*1024
    // f16 segments
    _Float16* f16base = (_Float16*)(vs + (size_t)B_ * D_);
    _Float16* sH  = f16base;            _Float16* sL  = sH  + Mr * D_;     // s split
    _Float16* hH  = sL  + Mr * D_;      _Float16* hL  = hH  + Mr * FF_;    // H split
    _Float16* pH  = hL  + Mr * FF_;     _Float16* pL  = pH  + (size_t)B_ * D_;
    _Float16* wvH = pL  + (size_t)B_ * D_;  _Float16* wvL = wvH + (size_t)D_ * D_;
    _Float16* w1H = wvL + (size_t)D_ * D_;  _Float16* w1L = w1H + (size_t)D_ * FF_;
    _Float16* w2H = w1L + (size_t)D_ * FF_; _Float16* w2L = w2H + (size_t)D_ * FF_;
    // total ws ~ 1.45 GB

    // 0) weight prep: split + transpose to [N][K] f16 hi/lo
    split_transpose_kernel<<<dim3(D_ / 32, D_ / 32),  256, 0, stream>>>(vw, wvH, wvL, D_, D_);
    split_transpose_kernel<<<dim3(FF_ / 32, D_ / 32), 256, 0, stream>>>(w1, w1H, w1L, D_, FF_);
    split_transpose_kernel<<<dim3(D_ / 32, FF_ / 32), 256, 0, stream>>>(w2, w2H, w2L, FF_, D_);

    // 1) piecesum over LEN2 (split)
    colsum_kernel<<<(B_ * D_) / 256, 256, 0, stream>>>(pieces, pH, pL);

    // 2) vsum = piecesum @ Wv + 64*bv  (softmax rowsum == 1 -> attn out == vsum)
    tdm_gemm_kernel<false, true, false><<<dim3(D_ / BN, B_ / BM), 256, 0, stream>>>(
        pH, pL, wvH, wvL, vb, (float)LEN2_, vs, nullptr, nullptr, B_, D_, D_);

    // 3) s = LN(state + vsum_broadcast, norm1)  (+ split copy for FF1)
    add_ln_kernel<<<(unsigned)Mr, 256, 0, stream>>>(state, vs, n1g, n1b, s, sH, sL, LEN1_, 1);

    // 4) H = relu(s @ W1 + b1) -> f16 hi/lo only
    tdm_gemm_kernel<true, false, true><<<dim3(FF_ / BN, (unsigned)(Mr / BM)), 256, 0, stream>>>(
        sH, sL, w1H, w1L, b1, 1.0f, nullptr, hH, hL, (int)Mr, FF_, D_);

    // 5) Y = H @ W2 + b2
    tdm_gemm_kernel<false, true, false><<<dim3(D_ / BN, (unsigned)(Mr / BM)), 256, 0, stream>>>(
        hH, hL, w2H, w2L, b2, 1.0f, Y, nullptr, nullptr, (int)Mr, D_, FF_);

    // 6) out = LN(s + Y, norm2)
    add_ln_kernel<<<(unsigned)Mr, 256, 0, stream>>>(s, Y, n2g, n2b, out, nullptr, nullptr, 1, 0);
}
